// AttentionHead_919123001963
// MI455X (gfx1250) — matrix operand order repaired and checked by reading, not compiled, not run on hardware
//
#include <hip/hip_runtime.h>

// Causal attention head, B=4 S=2048 E=1024, fp32 I/O.
// bf16 WMMA (v_wmma_f32_16x16x32_bf16) everywhere, fp32 accumulation/softmax.
//  Kernel 0: one-time f32->bf16 conversion of x and W_Q/W_K/W_V.
//  Kernel 1: fused QKV projection, 32x32 tile per wave per output;
//            V is written TRANSPOSED (V^T[b][e][s]) via per-wave LDS bounce.
//  Kernel 2: flash attention, one workgroup (8 waves) per 32-query tile;
//            Q tile staged to LDS with GLOBAL_LOAD_ASYNC_TO_LDS_B128,
//            P*V reads V^T directly from L2 (two 16B loads/lane), no staging.

#define BB 4
#define SS 2048
#define EE 1024

typedef __bf16 bf16_t;
typedef __attribute__((ext_vector_type(16))) __bf16 v16bf;
typedef __attribute__((ext_vector_type(8)))  __bf16 v8bf;
typedef __attribute__((ext_vector_type(4)))  __bf16 v4bf;
typedef __attribute__((ext_vector_type(8)))  float  v8f;

__device__ __forceinline__ v8f wmma_bf16(v16bf a, v16bf b, v8f c) {
  return __builtin_amdgcn_wmma_f32_16x16x32_bf16(false, a, false, b, (short)0, c,
                                                 false, false);
}

// ---- fragment loaders (layouts per CDNA5 ISA 7.12.2) ----
// A 16x32 (MxK), row-major source, leading dim ld (elements).
// lane<16: M=lane, K={0..7,16..23}; lane>=16: M=lane-16, K={8..15,24..31}
__device__ __forceinline__ v16bf load_a_bf16(const bf16_t* base, int ld) {
  int lane = threadIdx.x & 31;
  int m = lane & 15, h = lane >> 4;
  const bf16_t* p = base + m * ld + 8 * h;
  v8bf lo = *(const v8bf*)(p);
  v8bf hi = *(const v8bf*)(p + 16);
  v16bf a;
#pragma unroll
  for (int i = 0; i < 8; ++i) { a[i] = lo[i]; a[i + 8] = hi[i]; }
  return a;
}

// B 32x16 (KxN), source n-major: elem(k,n) at base[n*ld + k].
// lane<16: N=lane, K={0..7,16..23}; lane>=16: N=lane-16, K={8..15,24..31}
__device__ __forceinline__ v16bf load_b_bf16(const bf16_t* base, int ld) {
  int lane = threadIdx.x & 31;
  int n = lane & 15, h = lane >> 4;
  const bf16_t* p = base + n * ld + 8 * h;
  v8bf lo = *(const v8bf*)(p);
  v8bf hi = *(const v8bf*)(p + 16);
  v16bf b;
#pragma unroll
  for (int i = 0; i < 8; ++i) { b[i] = lo[i]; b[i + 8] = hi[i]; }
  return b;
}

// ---------------- kernel 0: f32 -> bf16 bulk convert ----------------
__global__ __launch_bounds__(256) void cvt_bf16_kernel(const float* __restrict__ src,
                                                       bf16_t* __restrict__ dst,
                                                       int n4) {
  int i = blockIdx.x * 256 + threadIdx.x;
  if (i < n4) {
    float4 v = ((const float4*)src)[i];
    v4bf o;
    o[0] = (bf16_t)v.x; o[1] = (bf16_t)v.y; o[2] = (bf16_t)v.z; o[3] = (bf16_t)v.w;
    ((v4bf*)dst)[i] = o;
  }
}

// ---------------- kernel 1: fused QKV projection ----------------
// y = x @ W^T : C[m=s, n=f] = sum_e x[s,e]*W[f,e]; B-frag reads W row-major.
__global__ __launch_bounds__(256) void qkv_proj_kernel(
    const bf16_t* __restrict__ xb, const bf16_t* __restrict__ wq,
    const bf16_t* __restrict__ wk, const bf16_t* __restrict__ wv,
    bf16_t* __restrict__ Qb, bf16_t* __restrict__ Kb, bf16_t* __restrict__ VT) {
  __shared__ __attribute__((aligned(16))) bf16_t TrV[8][32 * 40];  // 20 KB

  int w = threadIdx.x >> 5;
  int wid = (int)blockIdx.x * 8 + w;
  int mt = wid >> 5;   // 0..255 : 32-row tiles over B*S
  int ft = wid & 31;   // 0..31  : 32-col tiles over E
  int lane = threadIdx.x & 31;
  int n = lane & 15, h = lane >> 4;

  const bf16_t* wp[3] = {wq, wk, wv};
  v8f c[3][2][2];
#pragma unroll
  for (int g = 0; g < 3; ++g)
#pragma unroll
    for (int mi = 0; mi < 2; ++mi)
#pragma unroll
      for (int ni = 0; ni < 2; ++ni) c[g][mi][ni] = {};

  const bf16_t* xr = xb + (size_t)(mt * 32) * EE;
#pragma unroll 2
  for (int kk = 0; kk < EE / 32; ++kk) {
    v16bf a0 = load_a_bf16(xr + kk * 32, EE);
    v16bf a1 = load_a_bf16(xr + (size_t)16 * EE + kk * 32, EE);
#pragma unroll
    for (int g = 0; g < 3; ++g) {
      const bf16_t* wb = wp[g] + (size_t)(ft * 32) * EE + kk * 32;
      v16bf b0 = load_b_bf16(wb, EE);
      v16bf b1 = load_b_bf16(wb + (size_t)16 * EE, EE);
      c[g][0][0] = wmma_bf16(a0, b0, c[g][0][0]);
      c[g][1][0] = wmma_bf16(a1, b0, c[g][1][0]);
      c[g][0][1] = wmma_bf16(a0, b1, c[g][0][1]);
      c[g][1][1] = wmma_bf16(a1, b1, c[g][1][1]);
    }
  }

  // Q, K: row-major bf16 stores
#pragma unroll
  for (int g = 0; g < 2; ++g) {
    bf16_t* dst = g ? Kb : Qb;
#pragma unroll
    for (int mi = 0; mi < 2; ++mi)
#pragma unroll
      for (int ni = 0; ni < 2; ++ni) {
        size_t base = (size_t)(mt * 32 + mi * 16 + 8 * h) * EE + ft * 32 + ni * 16 + n;
#pragma unroll
        for (int r = 0; r < 8; ++r) dst[base + (size_t)r * EE] = (bf16_t)c[g][mi][ni][r];
      }
  }

  // V: transpose 32x32 block through per-wave LDS, store V^T[b][e][s]
  bf16_t* T = &TrV[w][0];
#pragma unroll
  for (int mi = 0; mi < 2; ++mi)
#pragma unroll
    for (int ni = 0; ni < 2; ++ni)
#pragma unroll
      for (int r = 0; r < 8; ++r)
        T[(ni * 16 + n) * 40 + mi * 16 + 8 * h + r] = (bf16_t)c[2][mi][ni][r];
  __syncthreads();
  {
    int bb = (mt * 32) >> 11;          // batch
    int s0 = (mt * 32) & (SS - 1);     // sequence offset within batch
    const v8bf* srcv = (const v8bf*)(T + lane * 40);   // column `lane`, 32 rows
    v8bf* dstv = (v8bf*)(VT + ((size_t)bb * EE + ft * 32 + lane) * SS + s0);
#pragma unroll
    for (int i = 0; i < 4; ++i) dstv[i] = srcv[i];
  }
}

// ---------------- kernel 2: flash attention (32-query tiles) ----------------
__global__ __launch_bounds__(256) void flash_attn_kernel(
    const bf16_t* __restrict__ Qb, const bf16_t* __restrict__ Kb,
    const bf16_t* __restrict__ VT, float* __restrict__ out) {
  __shared__ __attribute__((aligned(16))) bf16_t Qs[32 * EE];   // 64 KB query tile
  __shared__ __attribute__((aligned(16))) bf16_t Ps[32 * 256];  // 16 KB probs block
  __shared__ float red[8 * 32];
  __shared__ float rowm[32], rowl[32], rsc[32];

  const int tid = threadIdx.x;
  const int w = tid >> 5;
  const int lane = tid & 31;
  const int n = lane & 15, h = lane >> 4;

  const int b  = (int)blockIdx.x >> 6;          // / (S/32)
  const int q0 = ((int)blockIdx.x & 63) << 5;   // first query row of tile

  const bf16_t* Qg = Qb + ((size_t)b * SS + q0) * EE;
  const bf16_t* Kg = Kb + (size_t)b * SS * EE;
  const bf16_t* Vg = VT + (size_t)b * EE * SS + (size_t)(w * 128) * SS;

  // 32x1024 Q tile -> LDS. CDNA5 async copy (ASYNCcnt), plain copy fallback.
#if defined(__gfx1250__)
  for (int i = tid; i < 32 * EE / 8; i += 256) {
    unsigned lds_off = (unsigned)(unsigned long long)(Qs + i * 8);
    const bf16_t* g = Qg + (size_t)i * 8;
    asm volatile("global_load_async_to_lds_b128 %0, %1, off"
                 :: "v"(lds_off), "v"(g) : "memory");
  }
  asm volatile("s_wait_asynccnt 0" ::: "memory");
#else
  {
    const uint4* src = (const uint4*)Qg;
    uint4* dst = (uint4*)Qs;
    for (int i = tid; i < 32 * EE * 2 / 16; i += 256) dst[i] = src[i];
  }
#endif
  if (tid < 32) { rowm[tid] = -1e30f; rowl[tid] = 0.f; }
  __syncthreads();

  v8f O[2][8];  // 32x128 fp32 accumulator slice (cols w*128..+128)
#pragma unroll
  for (int qi = 0; qi < 2; ++qi)
#pragma unroll
    for (int t = 0; t < 8; ++t) O[qi][t] = {};

  const int nkb = (q0 + 32 + 255) >> 8;  // causal: only needed 256-key blocks
  for (int kb = 0; kb < nkb; ++kb) {
    const int kb0 = kb << 8;
    const int ks0 = kb0 + w * 32;  // this wave's 32 keys

    // 1) scores for this wave's 32 keys x 32 queries (skip fully masked)
    v8f s[2][2];
#pragma unroll
    for (int qi = 0; qi < 2; ++qi) { s[qi][0] = {}; s[qi][1] = {}; }
    if (ks0 <= q0 + 31) {
      const bf16_t* K0 = Kg + (size_t)ks0 * EE;
#pragma unroll 2
      for (int kk = 0; kk < EE / 32; ++kk) {
        v16bf b0 = load_b_bf16(K0 + kk * 32, EE);
        v16bf b1 = load_b_bf16(K0 + (size_t)16 * EE + kk * 32, EE);
        v16bf a0 = load_a_bf16(Qs + kk * 32, EE);
        v16bf a1 = load_a_bf16(Qs + (size_t)16 * EE + kk * 32, EE);
        s[0][0] = wmma_bf16(a0, b0, s[0][0]);
        s[0][1] = wmma_bf16(a0, b1, s[0][1]);
        s[1][0] = wmma_bf16(a1, b0, s[1][0]);
        s[1][1] = wmma_bf16(a1, b1, s[1][1]);
      }
    }
    // scale + causal mask + per-row max
    float rmax[2][8];
#pragma unroll
    for (int qi = 0; qi < 2; ++qi)
#pragma unroll
      for (int r = 0; r < 8; ++r) {
        int q = q0 + qi * 16 + r + 8 * h;
        float a0 = (ks0 + n      <= q) ? s[qi][0][r] * 0.03125f : -1e30f;
        float a1 = (ks0 + 16 + n <= q) ? s[qi][1][r] * 0.03125f : -1e30f;
        s[qi][0][r] = a0; s[qi][1][r] = a1;
        rmax[qi][r] = fmaxf(a0, a1);
      }
#pragma unroll
    for (int m = 1; m < 16; m <<= 1)
#pragma unroll
      for (int qi = 0; qi < 2; ++qi)
#pragma unroll
        for (int r = 0; r < 8; ++r)
          rmax[qi][r] = fmaxf(rmax[qi][r], __shfl_xor(rmax[qi][r], m, 32));
    if (n == 0) {  // lanes 0 / 16 publish rows 0-7 / 8-15 of each qi
#pragma unroll
      for (int qi = 0; qi < 2; ++qi)
#pragma unroll
        for (int r = 0; r < 8; ++r) red[w * 32 + qi * 16 + 8 * h + r] = rmax[qi][r];
    }
    __syncthreads();

    // 2) merge running softmax stats (one thread per query row)
    if (tid < 32) {
      float bm = red[tid];
#pragma unroll
      for (int ww = 1; ww < 8; ++ww) bm = fmaxf(bm, red[ww * 32 + tid]);
      float mold = rowm[tid];
      float mnew = fmaxf(mold, bm);
      float rs = __builtin_amdgcn_exp2f((mold - mnew) * 1.44269504f);
      rowm[tid] = mnew; rsc[tid] = rs; rowl[tid] *= rs;
    }
    __syncthreads();

    // 3) P = exp(s-m) -> Ps; rescale O; row sums
    float rsum[2][8];
#pragma unroll
    for (int qi = 0; qi < 2; ++qi)
#pragma unroll
      for (int r = 0; r < 8; ++r) {
        int row = qi * 16 + r + 8 * h;
        float mnew = rowm[row];
        float p0 = __builtin_amdgcn_exp2f((s[qi][0][r] - mnew) * 1.44269504f);
        float p1 = __builtin_amdgcn_exp2f((s[qi][1][r] - mnew) * 1.44269504f);
        Ps[row * 256 + w * 32 + n]      = (bf16_t)p0;
        Ps[row * 256 + w * 32 + 16 + n] = (bf16_t)p1;
        rsum[qi][r] = p0 + p1;
        float rs = rsc[row];
#pragma unroll
        for (int t = 0; t < 8; ++t) O[qi][t][r] *= rs;
      }
#pragma unroll
    for (int m = 1; m < 16; m <<= 1)
#pragma unroll
      for (int qi = 0; qi < 2; ++qi)
#pragma unroll
        for (int r = 0; r < 8; ++r) rsum[qi][r] += __shfl_xor(rsum[qi][r], m, 32);
    if (n == 0) {
#pragma unroll
      for (int qi = 0; qi < 2; ++qi)
#pragma unroll
        for (int r = 0; r < 8; ++r) red[w * 32 + qi * 16 + 8 * h + r] = rsum[qi][r];
    }
    __syncthreads();
    if (tid < 32) {
      float sum = 0.f;
#pragma unroll
      for (int ww = 0; ww < 8; ++ww) sum += red[ww * 32 + tid];
      rowl[tid] += sum;
    }
    __syncthreads();

    // 4) O += P(32x256) * V^T slice; each V B-frag feeds two P A-frags
    const int kkend = min(8, (q0 + 32 - kb0 + 31) >> 5);
    for (int kk = 0; kk < kkend; ++kk) {
      v16bf ap0 = load_a_bf16(Ps + kk * 32, 256);
      v16bf ap1 = load_a_bf16(Ps + 16 * 256 + kk * 32, 256);
#pragma unroll
      for (int t = 0; t < 8; ++t) {
        v16bf bv = load_b_bf16(Vg + (size_t)(t * 16) * SS + kb0 + kk * 32, SS);
        O[0][t] = wmma_bf16(ap0, bv, O[0][t]);
        O[1][t] = wmma_bf16(ap1, bv, O[1][t]);
      }
    }
    // next iteration's first barrier orders these Ps reads vs the next Ps writes
  }

  // epilogue: normalize by row sum, store fp32
  float* og = out + ((size_t)b * SS + q0) * EE + w * 128;
#pragma unroll
  for (int qi = 0; qi < 2; ++qi)
#pragma unroll
    for (int r = 0; r < 8; ++r) {
      int row = qi * 16 + r + 8 * h;
      float inv = 1.f / rowl[row];
#pragma unroll
      for (int t = 0; t < 8; ++t)
        og[(size_t)row * EE + t * 16 + n] = O[qi][t][r] * inv;
    }
}

extern "C" void kernel_launch(void* const* d_in, const int* in_sizes, int n_in,
                              void* d_out, int out_size, void* d_ws, size_t ws_size,
                              hipStream_t stream) {
  const float* x  = (const float*)d_in[0];
  const float* wq = (const float*)d_in[1];
  const float* wk = (const float*)d_in[2];
  const float* wv = (const float*)d_in[3];
  const size_t BSE = (size_t)BB * SS * EE;   // 8,388,608
  const size_t WSZ = (size_t)EE * EE;        // 1,048,576

  bf16_t* p   = (bf16_t*)d_ws;
  bf16_t* xb  = p; p += BSE;
  bf16_t* wqb = p; p += WSZ;
  bf16_t* wkb = p; p += WSZ;
  bf16_t* wvb = p; p += WSZ;
  bf16_t* Qb  = p; p += BSE;
  bf16_t* Kb  = p; p += BSE;
  bf16_t* VT  = p; p += BSE;                 // ~73 MB total scratch
  float* out = (float*)d_out;

  cvt_bf16_kernel<<<(int)(BSE / 4 / 256), 256, 0, stream>>>(x, xb, (int)(BSE / 4));
  cvt_bf16_kernel<<<(int)(WSZ / 4 / 256), 256, 0, stream>>>(wq, wqb, (int)(WSZ / 4));
  cvt_bf16_kernel<<<(int)(WSZ / 4 / 256), 256, 0, stream>>>(wk, wkb, (int)(WSZ / 4));
  cvt_bf16_kernel<<<(int)(WSZ / 4 / 256), 256, 0, stream>>>(wv, wvb, (int)(WSZ / 4));

  // 256 row-tiles * 32 col-tiles = 8192 waves / 8 per block
  qkv_proj_kernel<<<1024, 256, 0, stream>>>(xb, wqb, wkb, wvb, Qb, Kb, VT);
  // one workgroup per 32-query tile
  flash_attn_kernel<<<BB * SS / 32, 256, 0, stream>>>(Qb, Kb, VT, out);
}
